// TtMambaSSM_81063212745055
// MI455X (gfx1250) — compile-verified, matrix operand order checked by reading, not run on hardware
//
#include <hip/hip_runtime.h>
#include <math.h>

#define L_DIM 512
#define E_DIM 5120
#define R_DIM 160
#define N_DIM 16
#define NCAT  192   // 160 (dt_low) + 16 (B) + 16 (C)
#define CCH   8     // scan chunks over L
#define TCH   (L_DIM / CCH)  // 64 timesteps per chunk

typedef __attribute__((ext_vector_type(16))) __bf16       v16bf;
typedef __attribute__((ext_vector_type(8)))  float        v8f;
typedef __attribute__((ext_vector_type(8)))  unsigned int v8u;

__device__ __forceinline__ unsigned short f2bf(float f) {
  unsigned int u = __builtin_bit_cast(unsigned int, f);
  u += 0x7FFFu + ((u >> 16) & 1u);   // round-to-nearest-even
  return (unsigned short)(u >> 16);
}

// 16x32 bf16 fragment, row-major source [rows, ld] (bf16 stored as ushort).
// Per ISA 7.12.2: lanes 0-15 row=lane, VGPR0..3 -> K=0..7, VGPR4..7 -> K=16..23;
// lanes 16-31 same rows, K offset +8. B-matrix mirrors this, so the same loader
// works on a transposed (column-major) weight copy.
__device__ __forceinline__ v16bf load_frag16x32(const unsigned short* base, int ld,
                                                int row0, int k0) {
  int lane = threadIdx.x & 31;
  int r    = lane & 15;
  int hi   = lane >> 4;
  const unsigned short* p = base + (size_t)(row0 + r) * ld + k0 + 8 * hi;
  v8u u;
#pragma unroll
  for (int v = 0; v < 4; ++v) u[v]     = *(const unsigned int*)(p + 2 * v);
#pragma unroll
  for (int v = 0; v < 4; ++v) u[4 + v] = *(const unsigned int*)(p + 16 + 2 * v);
  return __builtin_bit_cast(v16bf, u);
}

// ---------------------------------------------------------------------------
// Pack: x -> bf16; WcatT [NCAT, E] = [W_dt_low | W_B | W_C]^T bf16;
// WdtT [E, R] = W_dt^T bf16.
// ---------------------------------------------------------------------------
__global__ __launch_bounds__(256) void pack_kernel(
    const float* __restrict__ x, const float* __restrict__ Wlow,
    const float* __restrict__ Wdt, const float* __restrict__ WB,
    const float* __restrict__ WC, unsigned short* __restrict__ xbf,
    unsigned short* __restrict__ WcatT, unsigned short* __restrict__ WdtT) {
  const int LE   = L_DIM * E_DIM;
  const int WCAT = NCAT * E_DIM;
  const int WDTT = E_DIM * R_DIM;
  int idx = blockIdx.x * blockDim.x + threadIdx.x;
  if (idx < LE) {
    xbf[idx] = f2bf(x[idx]);
  } else if (idx < LE + WCAT) {
    int i = idx - LE;
    int n = i / E_DIM;
    int k = i - n * E_DIM;
    float v;
    if (n < R_DIM)              v = Wlow[(size_t)k * R_DIM + n];
    else if (n < R_DIM + N_DIM) v = WB[(size_t)k * N_DIM + (n - R_DIM)];
    else                        v = WC[(size_t)k * N_DIM + (n - R_DIM - N_DIM)];
    WcatT[i] = f2bf(v);
  } else if (idx < LE + WCAT + WDTT) {
    int i = idx - LE - WCAT;
    int n = i / R_DIM;
    int k = i - n * R_DIM;
    WdtT[i] = f2bf(Wdt[(size_t)k * E_DIM + n]);
  }
}

// ---------------------------------------------------------------------------
// GEMM1: x[L,E]bf16 @ [W_dt_low|W_B|W_C] -> dt_low bf16 + B,C f32.
// One wave per 16x16 tile, K=5120 in steps of 32.
// ---------------------------------------------------------------------------
__global__ __launch_bounds__(256) void gemm1_kernel(
    const unsigned short* __restrict__ xbf, const unsigned short* __restrict__ WcatT,
    unsigned short* __restrict__ T1bf, float* __restrict__ Bm, float* __restrict__ Cm) {
  int wave = blockIdx.x * (blockDim.x >> 5) + (threadIdx.x >> 5);
  const int ntn = NCAT / 16;             // 12 tiles in N
  int tm = wave / ntn;
  int tn = wave - tm * ntn;
  int row0 = tm * 16, n0 = tn * 16;
  v8f acc = {};
  for (int k0 = 0; k0 < E_DIM; k0 += 32) {
    v16bf a = load_frag16x32(xbf,   E_DIM, row0, k0);
    v16bf b = load_frag16x32(WcatT, E_DIM, n0,   k0);
    acc = __builtin_amdgcn_wmma_f32_16x16x32_bf16(false, a, false, b,
                                                  (short)0, acc, false, false);
  }
  int lane = threadIdx.x & 31;
  int nn = lane & 15, hi = lane >> 4;
  int col = n0 + nn;
#pragma unroll
  for (int r = 0; r < 8; ++r) {
    int row = row0 + r + 8 * hi;
    float v = acc[r];
    if (col < R_DIM)              T1bf[row * R_DIM + col] = f2bf(v);
    else if (col < R_DIM + N_DIM) Bm[row * N_DIM + (col - R_DIM)] = v;
    else                          Cm[row * N_DIM + (col - R_DIM - N_DIM)] = v;
  }
}

// ---------------------------------------------------------------------------
// GEMM2: dt_low[L,R]bf16 @ W_dt[R,E] + bias -> thresholded softplus -> delta f32.
// ---------------------------------------------------------------------------
__global__ __launch_bounds__(256) void gemm2_kernel(
    const unsigned short* __restrict__ T1bf, const unsigned short* __restrict__ WdtT,
    const float* __restrict__ dt_bias, float* __restrict__ delta) {
  int wave = blockIdx.x * (blockDim.x >> 5) + (threadIdx.x >> 5);
  const int ntn = E_DIM / 16;            // 320 tiles in N
  int tm = wave / ntn;
  int tn = wave - tm * ntn;
  int row0 = tm * 16, n0 = tn * 16;
  v8f acc = {};
#pragma unroll
  for (int k0 = 0; k0 < R_DIM; k0 += 32) {
    v16bf a = load_frag16x32(T1bf, R_DIM, row0, k0);
    v16bf b = load_frag16x32(WdtT, R_DIM, n0,   k0);
    acc = __builtin_amdgcn_wmma_f32_16x16x32_bf16(false, a, false, b,
                                                  (short)0, acc, false, false);
  }
  int lane = threadIdx.x & 31;
  int nn = lane & 15, hi = lane >> 4;
  int col = n0 + nn;
  float bias = dt_bias[col];
#pragma unroll
  for (int r = 0; r < 8; ++r) {
    int row = row0 + r + 8 * hi;
    float z = acc[r] + bias;
    delta[(size_t)row * E_DIM + col] = (z > 20.0f) ? z : log1pf(__expf(z));
  }
}

// ---------------------------------------------------------------------------
// Scan pass 1: per (chunk c, channel e): chunk-local scan from h=0 plus the
// running product of abar over the chunk. 8x the parallelism of a full-L scan.
// ---------------------------------------------------------------------------
__global__ __launch_bounds__(256) void scan_p1_kernel(
    const float* __restrict__ x, const float* __restrict__ delta,
    const float* __restrict__ Bm, const float* __restrict__ A_log,
    float* __restrict__ Acum, float* __restrict__ Hloc) {
  __shared__ float sB[TCH * N_DIM];      // 4 KB
  int e = blockIdx.x * blockDim.x + threadIdx.x;
  int c = blockIdx.y;
  float a[N_DIM], p[N_DIM], h[N_DIM];
#pragma unroll
  for (int n = 0; n < N_DIM; ++n) {
    a[n] = -__expf(A_log[(size_t)e * N_DIM + n]);   // A = -exp(A_log)
    p[n] = 1.0f;
    h[n] = 0.0f;
  }
  const float* gB = Bm + (size_t)c * TCH * N_DIM;
  for (int i = threadIdx.x; i < TCH * N_DIM; i += blockDim.x) sB[i] = gB[i];
  __syncthreads();
  for (int tt = 0; tt < TCH; ++tt) {
    int t = c * TCH + tt;
    float dt  = delta[(size_t)t * E_DIM + e];
    float xv  = x[(size_t)t * E_DIM + e];
    float dbx = dt * xv;
#pragma unroll
    for (int n = 0; n < N_DIM; ++n) {
      float ab = __expf(dt * a[n]);                 // abar
      p[n] *= ab;
      h[n] = fmaf(ab, h[n], dbx * sB[tt * N_DIM + n]);
    }
  }
  size_t base = ((size_t)c * E_DIM + e) * N_DIM;
#pragma unroll
  for (int n = 0; n < N_DIM; ++n) {
    Acum[base + n] = p[n];
    Hloc[base + n] = h[n];
  }
}

// ---------------------------------------------------------------------------
// Scan pass 2: sequential combine across the 8 chunks -> entry state per chunk.
// One thread per (e,n) state; fully coalesced.
// ---------------------------------------------------------------------------
__global__ __launch_bounds__(256) void scan_p2_kernel(
    const float* __restrict__ Acum, const float* __restrict__ Hloc,
    float* __restrict__ Hinit) {
  int idx = blockIdx.x * blockDim.x + threadIdx.x;   // e*N + n
  float h = 0.0f;
#pragma unroll
  for (int c = 0; c < CCH; ++c) {
    size_t o = (size_t)c * E_DIM * N_DIM + idx;
    Hinit[o] = h;
    h = fmaf(Acum[o], h, Hloc[o]);
  }
}

// ---------------------------------------------------------------------------
// Scan pass 3: re-run each chunk from its entry state, fused with
// y = sum_n h*C and out = x*D + y. abar is recomputed (cheaper than a
// 168 MB h/abar round-trip through HBM).
// ---------------------------------------------------------------------------
__global__ __launch_bounds__(256) void scan_p3_kernel(
    const float* __restrict__ x, const float* __restrict__ delta,
    const float* __restrict__ Bm, const float* __restrict__ Cm,
    const float* __restrict__ A_log, const float* __restrict__ Dv,
    const float* __restrict__ Hinit, float* __restrict__ out) {
  __shared__ float sB[TCH * N_DIM];
  __shared__ float sC[TCH * N_DIM];
  int e = blockIdx.x * blockDim.x + threadIdx.x;
  int c = blockIdx.y;
  float a[N_DIM], h[N_DIM];
  size_t base = ((size_t)c * E_DIM + e) * N_DIM;
#pragma unroll
  for (int n = 0; n < N_DIM; ++n) {
    a[n] = -__expf(A_log[(size_t)e * N_DIM + n]);
    h[n] = Hinit[base + n];
  }
  float dch = Dv[e];
  const float* gB = Bm + (size_t)c * TCH * N_DIM;
  const float* gC = Cm + (size_t)c * TCH * N_DIM;
  for (int i = threadIdx.x; i < TCH * N_DIM; i += blockDim.x) {
    sB[i] = gB[i];
    sC[i] = gC[i];
  }
  __syncthreads();
  for (int tt = 0; tt < TCH; ++tt) {
    int t = c * TCH + tt;
    float dt  = delta[(size_t)t * E_DIM + e];
    float xv  = x[(size_t)t * E_DIM + e];
    float dbx = dt * xv;
    float y = 0.0f;
#pragma unroll
    for (int n = 0; n < N_DIM; ++n) {
      float ab = __expf(dt * a[n]);
      h[n] = fmaf(ab, h[n], dbx * sB[tt * N_DIM + n]);
      y    = fmaf(h[n], sC[tt * N_DIM + n], y);
    }
    out[(size_t)t * E_DIM + e] = fmaf(xv, dch, y);
  }
}

extern "C" void kernel_launch(void* const* d_in, const int* in_sizes, int n_in,
                              void* d_out, int out_size, void* d_ws, size_t ws_size,
                              hipStream_t stream) {
  const float* x     = (const float*)d_in[0];
  const float* Wlow  = (const float*)d_in[1];
  const float* Wdt   = (const float*)d_in[2];
  const float* dtb   = (const float*)d_in[3];
  const float* WB    = (const float*)d_in[4];
  const float* WC    = (const float*)d_in[5];
  const float* A_log = (const float*)d_in[6];
  const float* Dv    = (const float*)d_in[7];
  float* out = (float*)d_out;

  char* ws = (char*)d_ws;
  size_t off = 0;
  auto alloc = [&](size_t bytes) -> void* {
    void* p = ws + off;
    off += (bytes + 255) & ~(size_t)255;
    return p;
  };
  unsigned short* xbf   = (unsigned short*)alloc((size_t)L_DIM * E_DIM * 2);
  unsigned short* WcatT = (unsigned short*)alloc((size_t)NCAT * E_DIM * 2);
  unsigned short* WdtT  = (unsigned short*)alloc((size_t)E_DIM * R_DIM * 2);
  unsigned short* T1bf  = (unsigned short*)alloc((size_t)L_DIM * R_DIM * 2);
  float* Bm    = (float*)alloc((size_t)L_DIM * N_DIM * 4);
  float* Cm    = (float*)alloc((size_t)L_DIM * N_DIM * 4);
  float* delta = (float*)alloc((size_t)L_DIM * E_DIM * 4);
  float* Acum  = (float*)alloc((size_t)CCH * E_DIM * N_DIM * 4);
  float* Hloc  = (float*)alloc((size_t)CCH * E_DIM * N_DIM * 4);
  float* Hinit = (float*)alloc((size_t)CCH * E_DIM * N_DIM * 4);

  int pack_total = L_DIM * E_DIM + NCAT * E_DIM + E_DIM * R_DIM;
  pack_kernel<<<(pack_total + 255) / 256, 256, 0, stream>>>(
      x, Wlow, Wdt, WB, WC, xbf, WcatT, WdtT);

  gemm1_kernel<<<(L_DIM / 16) * (NCAT / 16) / 8, 256, 0, stream>>>(
      xbf, WcatT, T1bf, Bm, Cm);

  gemm2_kernel<<<(L_DIM / 16) * (E_DIM / 16) / 8, 256, 0, stream>>>(
      T1bf, WdtT, dtb, delta);

  dim3 sg(E_DIM / 256, CCH);
  scan_p1_kernel<<<sg, 256, 0, stream>>>(x, delta, Bm, A_log, Acum, Hloc);
  scan_p2_kernel<<<(E_DIM * N_DIM) / 256, 256, 0, stream>>>(Acum, Hloc, Hinit);
  scan_p3_kernel<<<sg, 256, 0, stream>>>(x, delta, Bm, Cm, A_log, Dv, Hinit, out);
}
